// CRF_59493886984489
// MI455X (gfx1250) — matrix operand order repaired and checked by reading, not compiled, not run on hardware
//
#include <hip/hip_runtime.h>
#include <hip/hip_bf16.h>

// CRF NLL on MI455X (gfx1250), wave32 + WMMA.
//
// Forward scan in scaled-linear domain:
//   p_t = exp(score_t - c_t),  q = p × exp(trans)  (WMMA f16->f32),
//   u = q * exp(emit_t),  m = max(u),  c += log(m),  p' = u/m.
// A-operand = exp(trans)^T held constant in 6 v16h fragments; B-operand = p,
// repacked per step from the WMMA D layout purely in-register (one lane
// half-swap via shfl_xor(16)), so the 512-step chain has no LDS transpose.

typedef __attribute__((ext_vector_type(16))) _Float16 v16h;
typedef __attribute__((ext_vector_type(8)))  float    v8f;

#define CRF_B 1024
#define CRF_T 512
#define CRF_K 48

__global__ __launch_bounds__(32) void crf_forward_wmma(
    const float* __restrict__ em,      // (B,T,K)
    const float* __restrict__ trans,   // (K,K)
    const float* __restrict__ startt,  // (K)
    const float* __restrict__ endt,    // (K)
    float* __restrict__ logz)          // (B)
{
  const int lane = threadIdx.x & 31;
  const int bn   = lane & 15;   // batch column within 16-row tile (D/B "N")
  const int hi   = lane >> 4;   // half-wave: D rows 0-7 vs 8-15
  const int b    = blockIdx.x * 16 + bn;
  const float* erow = em + (size_t)b * CRF_T * CRF_K;

  // ---- A fragments: A[m][k] = exp(trans[i][j]), i = ck*32 + K(h,lane),
  // j = 16n + (lane%16). 16-bit A layout (contiguous, matches B): K=h;
  // lanes>=16 add 16. Zero-pad i >= 48. Constant across the whole t-loop.
  v16h aE[3][2];
#pragma unroll
  for (int n = 0; n < 3; ++n)
#pragma unroll
    for (int ck = 0; ck < 2; ++ck)
#pragma unroll
      for (int h = 0; h < 16; ++h) {
        const int Kh = h + (hi ? 16 : 0);
        const int i  = ck * 32 + Kh;
        const int j  = n * 16 + bn;
        const float v = (i < CRF_K) ? __expf(trans[i * CRF_K + j]) : 0.0f;
        aE[n][ck][h] = (_Float16)v;
      }

  // u[n][v] = unnormalized p for state j = 16n + 8*hi + v (D layout).
  float u[3][8];
  float c = 0.0f, rinv = 1.0f;
  v16h bq0 = {}, bq1 = {};  // B fragments: states 0-31 and 32-63 (48-63 pad=0)

  // D layout -> B layout repack + renormalization.
  // B k order used here (f16, 32x16, contiguous as in A): lanes<16 hold rows K=0..15, lanes>=16 rows K=16..31
  // across VGPR halves h=0..15. State s lives in u[s/16][s%8] of half-wave
  // (s%16)/8, so the off-half values come from one shfl_xor(16).
  auto renorm_pack = [&]() {
    float m = u[0][0];
#pragma unroll
    for (int n = 0; n < 3; ++n)
#pragma unroll
      for (int v = 0; v < 8; ++v) m = fmaxf(m, u[n][v]);
    m = fmaxf(m, __shfl_xor(m, 16, 32));
    c += __logf(m);
    rinv = __builtin_amdgcn_rcpf(m);
    float us[3][8];
#pragma unroll
    for (int n = 0; n < 3; ++n)
#pragma unroll
      for (int v = 0; v < 8; ++v) us[n][v] = __shfl_xor(u[n][v], 16, 32);
#pragma unroll
    for (int h = 0; h < 16; ++h) {
      // chunk0: state s = 16*hi + h ; chunk1: s = 32 + 16*hi + h (>=48 pad)
      const float a0 = (h < 8) ? (hi ? us[1][h] : u[0][h])
                               : (hi ? u[1][h - 8] : us[0][h - 8]);
      const float a1 = hi ? 0.0f : ((h < 8) ? u[2][h] : us[2][h - 8]);
      bq0[h] = (_Float16)(a0 * rinv);
      bq1[h] = (_Float16)(a1 * rinv);
    }
  };

  // ---- t = 0: p0 = exp(start + emissions[:,0])
#pragma unroll
  for (int n = 0; n < 3; ++n) {
    const float4* p4 = (const float4*)(erow + n * 16 + hi * 8);
    const float4 e0 = p4[0], e1 = p4[1];
    const float ev[8] = {e0.x, e0.y, e0.z, e0.w, e1.x, e1.y, e1.z, e1.w};
#pragma unroll
    for (int v = 0; v < 8; ++v) {
      const int j = n * 16 + hi * 8 + v;
      u[n][v] = __expf(startt[j] + ev[v]);
    }
  }
  renorm_pack();

  // ---- sequential scan: q = p x exp(trans) via 6 WMMA per step
  for (int t = 1; t < CRF_T; ++t) {
    v8f d[3];
#pragma unroll
    for (int n = 0; n < 3; ++n) {
      v8f z = {};
      z = __builtin_amdgcn_wmma_f32_16x16x32_f16(false, aE[n][0], false, bq0,
                                                 (short)0, z, false, false);
      z = __builtin_amdgcn_wmma_f32_16x16x32_f16(false, aE[n][1], false, bq1,
                                                 (short)0, z, false, false);
      d[n] = z;
    }
    const float* et = erow + (size_t)t * CRF_K;
#pragma unroll
    for (int n = 0; n < 3; ++n) {
      const float4* p4 = (const float4*)(et + n * 16 + hi * 8);
      const float4 e0 = p4[0], e1 = p4[1];
      const float ev[8] = {e0.x, e0.y, e0.z, e0.w, e1.x, e1.y, e1.z, e1.w};
#pragma unroll
      for (int v = 0; v < 8; ++v) u[n][v] = d[n][v] * __expf(ev[v]);
    }
    renorm_pack();
  }

  // ---- log_Z = c + log( sum_j p[j]*exp(end[j]) )
  float s = 0.0f;
#pragma unroll
  for (int n = 0; n < 3; ++n)
#pragma unroll
    for (int v = 0; v < 8; ++v) {
      const int j = n * 16 + hi * 8 + v;
      s += u[n][v] * __expf(endt[j]);
    }
  s *= rinv;
  s += __shfl_xor(s, 16, 32);
  if (hi == 0) logz[b] = c + __logf(s);
}

// Gold path score: start[tag0] + emit[0,tag0] + sum_t (trans + emit) + end[last]
// (mask is all-true in this problem). One block per batch row.
__global__ __launch_bounds__(256) void crf_gold(
    const float* __restrict__ em, const int* __restrict__ tags,
    const float* __restrict__ trans, const float* __restrict__ startt,
    const float* __restrict__ endt, float* __restrict__ gold)
{
  const int b = blockIdx.x, tid = threadIdx.x;
  const float* erow = em + (size_t)b * CRF_T * CRF_K;
  const int* tg = tags + (size_t)b * CRF_T;
  __shared__ float red[256];
  float s = 0.0f;
  for (int t = tid; t < CRF_T; t += 256) {
    const int tag = tg[t];
    s += erow[(size_t)t * CRF_K + tag];
    if (t > 0) s += trans[tg[t - 1] * CRF_K + tag];
  }
  if (tid == 0) s += startt[tg[0]] + endt[tg[CRF_T - 1]];
  red[tid] = s;
  __syncthreads();
  for (int ofs = 128; ofs > 0; ofs >>= 1) {
    if (tid < ofs) red[tid] += red[tid + ofs];
    __syncthreads();
  }
  if (tid == 0) gold[b] = red[0];
}

// mean(log_Z - gold) with a single-block deterministic reduction.
__global__ __launch_bounds__(256) void crf_final(
    const float* __restrict__ logz, const float* __restrict__ gold,
    float* __restrict__ out)
{
  const int tid = threadIdx.x;
  __shared__ float red[256];
  float s = 0.0f;
  for (int i = tid; i < CRF_B; i += 256) s += logz[i] - gold[i];
  red[tid] = s;
  __syncthreads();
  for (int ofs = 128; ofs > 0; ofs >>= 1) {
    if (tid < ofs) red[tid] += red[tid + ofs];
    __syncthreads();
  }
  if (tid == 0) out[0] = red[0] * (1.0f / (float)CRF_B);
}

extern "C" void kernel_launch(void* const* d_in, const int* in_sizes, int n_in,
                              void* d_out, int out_size, void* d_ws, size_t ws_size,
                              hipStream_t stream) {
  const float* em    = (const float*)d_in[0];  // emissions (B,T,K) f32
  const int*   tags  = (const int*)d_in[1];    // tags (B,T) i32
  //            d_in[2] = mask (all true) — semantics folded in, unused
  const float* trans = (const float*)d_in[3];  // (K,K)
  const float* st    = (const float*)d_in[4];  // (K)
  const float* en    = (const float*)d_in[5];  // (K)

  float* logz = (float*)d_ws;
  float* gold = logz + CRF_B;

  crf_forward_wmma<<<CRF_B / 16, 32, 0, stream>>>(em, trans, st, en, logz);
  crf_gold<<<CRF_B, 256, 0, stream>>>(em, tags, trans, st, en, gold);
  crf_final<<<1, 256, 0, stream>>>(logz, gold, (float*)d_out);
}
